// MIGAModel_37237366456667
// MI455X (gfx1250) — matrix operand order repaired and checked by reading, not compiled, not run on hardware
//
#include <hip/hip_runtime.h>
#include <hip/hip_bf16.h>

#define N_SAMP 20000
#define KDIM   9480
#define GE_DIM 128
#define MT     80                        // rows per GEMM block (20000/80 = 250 blocks)
#define KC2    64                        // K per staged chunk (2 WMMA k-steps)
#define NCH    ((KDIM + KC2 - 1) / KC2)  // 149 chunks; last has 8 valid k
#define KT_PAD (NCH * KC2)               // 9536: zero-padded Wr^T row length
#define SPB    16                        // samples per epilogue block

typedef __attribute__((ext_vector_type(16))) __bf16 v16bf;
typedef __attribute__((ext_vector_type(8)))  float  v8f;

union BfFrag { v16bf v; unsigned short s[16]; uint4 u4[2]; };

__device__ __forceinline__ unsigned short f32_to_bf16(float f) {
  unsigned int u = __builtin_bit_cast(unsigned int, f);
  u += 0x7FFFu + ((u >> 16) & 1u);       // round-to-nearest-even
  return (unsigned short)(u >> 16);
}

__device__ __forceinline__ unsigned int pack_bf16x2(float lo, float hi) {
  return (unsigned int)f32_to_bf16(lo) | ((unsigned int)f32_to_bf16(hi) << 16);
}

// Async global->LDS copy (gfx1250): 16B per lane, tracked by ASYNCcnt.
__device__ __forceinline__ void async_copy_b128(unsigned lds_addr, const void* gaddr) {
  asm volatile("global_load_async_to_lds_b128 %0, %1, off"
               :: "v"(lds_addr), "v"(gaddr) : "memory");
}
// Drain XCNT: address translation of in-flight VMEM done -> compiler may
// safely recycle the address VGPRs our inline asm consumed (XNACK replay).
__device__ __forceinline__ void wait_xcnt0() {
  asm volatile("s_wait_xcnt 0x0" ::: "memory");
}
__device__ __forceinline__ void wait_async0() {
  asm volatile("s_wait_asynccnt 0x0" ::: "memory");
}

// ---------------------------------------------------------------------------
// Kernel 0 (one-shot): Wr[9480,128] -> bf16 Wr^T[128][KT_PAD], zero-padded K.
// Removes the transpose AND the K-tail guard from the router hot loop.
// ---------------------------------------------------------------------------
__global__ __launch_bounds__(256)
void miga_wr_transpose(const float* __restrict__ Wr, unsigned int* __restrict__ wrt)
{
  const int rowU = KT_PAD / 2;                       // 4768 uints per row
  int uidx = blockIdx.x * 256 + threadIdx.x;         // exact: 128*rowU/256 blocks
  int n  = uidx / rowU;
  int kt = (uidx - n * rowU) * 2;
  float lo = (kt     < KDIM) ? Wr[(size_t)kt * GE_DIM + n]       : 0.0f;
  float hi = (kt + 1 < KDIM) ? Wr[(size_t)(kt + 1) * GE_DIM + n] : 0.0f;
  wrt[uidx] = pack_bf16x2(lo, hi);
}

// ---------------------------------------------------------------------------
// Kernel A: router GEMM  h[N,128] = x[N,9480] @ Wr + br   (bf16 WMMA)
// 8 waves: wave w owns 16-col tile; 5 M-subtiles of 16 rows; LDS ping-pong.
// B staging via async global->LDS DMA (ASYNCcnt); A staged with f32->bf16.
// ---------------------------------------------------------------------------
__global__ __launch_bounds__(256)
void miga_router(const float* __restrict__ x, const unsigned short* __restrict__ wrt,
                 const float* __restrict__ br, float* __restrict__ hout)
{
  __shared__ unsigned short sA[2][MT][KC2];          // 2 x 10 KB
  __shared__ unsigned short sB[2][GE_DIM][KC2];      // 2 x 16 KB

  const int tid  = threadIdx.x;
  const int wave = tid >> 5;
  const int lane = tid & 31;
  const int m0   = blockIdx.x * MT;
  const int nLoc = lane & 15;
  const int hi   = lane >> 4;
  const int col  = wave * 16 + nLoc;

  v8f acc[5] = {};

  // B chunk: 128x64 bf16 (16 KB), already padded: pure async DMA, 4 x b128/thread.
  auto stageB = [&](int c, int buf) {
    const int kb = c * KC2;
    #pragma unroll
    for (int i = 0; i < 4; ++i) {
      int fi = tid + i * 256;
      int n = fi >> 3, qq = fi & 7;
      const void* g = wrt + (size_t)n * KT_PAD + kb + qq * 8;
      unsigned lds = (unsigned)(uintptr_t)&sB[buf][n][qq * 8];
      async_copy_b128(lds, g);
    }
    wait_xcnt0();   // translation done: address VGPRs may be recycled safely
  };

  // A chunk: 80x64 f32 -> bf16. 1280 float4 / 256 thr = 5 each, 16B aligned.
  auto stageA = [&](int c, int buf) {
    const int kb = c * KC2;
    #pragma unroll
    for (int i = 0; i < 5; ++i) {
      int fi = tid + i * 256;
      int m = fi >> 4, q = fi & 15;
      const float* src = x + (size_t)(m0 + m) * KDIM + kb + q * 4;
      float v0, v1, v2, v3;
      if (kb + KC2 <= KDIM) {                        // hot path: no guards
        float4 v = *(const float4*)src;
        v0 = v.x; v1 = v.y; v2 = v.z; v3 = v.w;
      } else {                                       // only chunk 148
        int k0 = kb + q * 4;
        v0 = (k0     < KDIM) ? src[0] : 0.0f;
        v1 = (k0 + 1 < KDIM) ? src[1] : 0.0f;
        v2 = (k0 + 2 < KDIM) ? src[2] : 0.0f;
        v3 = (k0 + 3 < KDIM) ? src[3] : 0.0f;
      }
      uint2 p; p.x = pack_bf16x2(v0, v1); p.y = pack_bf16x2(v2, v3);
      *(uint2*)&sA[buf][m][q * 4] = p;
    }
  };

  stageB(0, 0);
  stageA(0, 0);
  for (int c = 0; c < NCH; ++c) {
    const int buf = c & 1;
    wait_async0();                                   // our async B copy done
    __syncthreads();                                 // everyone's staging visible
    if (c + 1 < NCH) {                               // overlap with compute
      stageB(c + 1, buf ^ 1);                        // DMA in background
      stageA(c + 1, buf ^ 1);
    }

    #pragma unroll
    for (int ks = 0; ks < KC2; ks += 32) {
      // B frag: lane n=col; VGPR 0-3 = K ks+8hi..+7, VGPR 4-7 = K ks+16+8hi..+7
      BfFrag bfr;
      bfr.u4[0] = *(const uint4*)&sB[buf][col][ks + 8 * hi];
      bfr.u4[1] = *(const uint4*)&sB[buf][col][ks + 16 + 8 * hi];
      #pragma unroll
      for (int s5 = 0; s5 < 5; ++s5) {
        // A frag: VGPR 0-3 = K ks+8hi..+7, VGPR 4-7 = K ks+16+8hi..+7
        BfFrag af;
        const int m = s5 * 16 + nLoc;
        af.u4[0] = *(const uint4*)&sA[buf][m][ks + 8 * hi];
        af.u4[1] = *(const uint4*)&sA[buf][m][ks + 16 + 8 * hi];
        acc[s5] = __builtin_amdgcn_wmma_f32_16x16x32_bf16(
            false, af.v, false, bfr.v, (short)0, acc[s5], false, false);
      }
    }
  }

  const float bias = br[col];
  #pragma unroll
  for (int s5 = 0; s5 < 5; ++s5)
    #pragma unroll
    for (int r = 0; r < 8; ++r) {
      int m = m0 + s5 * 16 + hi * 8 + r;             // C layout: M = r + 8*hi
      hout[(size_t)m * GE_DIM + col] = acc[s5][r] + bias;
    }
}

// ---------------------------------------------------------------------------
// Kernel B: expert GEMM  eo[N,128] = h[N,128] @ We^T + be
// We is [128 experts][128 k] row-major -> already the [n][k] LDS layout.
// ---------------------------------------------------------------------------
__global__ __launch_bounds__(256)
void miga_experts(const float* __restrict__ h, const float* __restrict__ We,
                  const float* __restrict__ be, float* __restrict__ eo)
{
  __shared__ unsigned short sA[MT][GE_DIM];          // 80x128 bf16 (20 KB)
  __shared__ unsigned short sB[GE_DIM][GE_DIM];      // [f][k]     (32 KB)

  const int tid  = threadIdx.x;
  const int wave = tid >> 5;
  const int lane = tid & 31;
  const int m0   = blockIdx.x * MT;
  const int nLoc = lane & 15;
  const int hi   = lane >> 4;
  const int col  = wave * 16 + nLoc;

  #pragma unroll
  for (int i = 0; i < (MT * GE_DIM) / 512; ++i) {    // 2 floats -> 1 uint pack
    int idx = (tid + i * 256) * 2;
    float a = h[(size_t)m0 * GE_DIM + idx];
    float b = h[(size_t)m0 * GE_DIM + idx + 1];
    *(unsigned int*)&sA[idx >> 7][idx & 127] = pack_bf16x2(a, b);
  }
  #pragma unroll
  for (int i = 0; i < (GE_DIM * GE_DIM) / 512; ++i) {
    int idx = (tid + i * 256) * 2;
    float a = We[idx], b = We[idx + 1];
    *(unsigned int*)&sB[idx >> 7][idx & 127] = pack_bf16x2(a, b);
  }
  __syncthreads();

  v8f acc[5] = {};
  #pragma unroll
  for (int c = 0; c < GE_DIM / 32; ++c) {            // 4 K-steps
    int kb = c * 32;
    BfFrag bfr;
    bfr.u4[0] = *(const uint4*)&sB[col][kb + 8 * hi];
    bfr.u4[1] = *(const uint4*)&sB[col][kb + 16 + 8 * hi];
    #pragma unroll
    for (int s5 = 0; s5 < 5; ++s5) {
      BfFrag af;
      int m = s5 * 16 + nLoc;
      af.u4[0] = *(const uint4*)&sA[m][kb + 8 * hi];
      af.u4[1] = *(const uint4*)&sA[m][kb + 16 + 8 * hi];
      acc[s5] = __builtin_amdgcn_wmma_f32_16x16x32_bf16(
          false, af.v, false, bfr.v, (short)0, acc[s5], false, false);
    }
  }

  const float bias = be[col];
  #pragma unroll
  for (int s5 = 0; s5 < 5; ++s5)
    #pragma unroll
    for (int r = 0; r < 8; ++r) {
      int m = m0 + s5 * 16 + hi * 8 + r;
      eo[(size_t)m * GE_DIM + col] = acc[s5][r] + bias;
    }
}

// ---------------------------------------------------------------------------
// Kernel C: top-2 gating + per-group attention + output projection.
// 128 threads (t = g*16+e) handle one sample at a time; SPB samples per block.
// ---------------------------------------------------------------------------
__global__ __launch_bounds__(128)
void miga_attn(const float* __restrict__ h,  const float* __restrict__ eo,
               const float* __restrict__ Wq, const float* __restrict__ bq,
               const float* __restrict__ Wk, const float* __restrict__ bk,
               const float* __restrict__ Wv, const float* __restrict__ bv,
               const float* __restrict__ Wo, const float* __restrict__ bo,
               float* __restrict__ out)
{
  __shared__ float sWq[2048], sWk[2048], sWv[2048], sWo[2048];
  __shared__ float sbq[128], sbk[128], sbv[128], sbo[128];
  __shared__ float sh[128], seo[128], sq[128], sk[128], sv[128];
  __shared__ float ss[128], sat[128], saf[128], so[128];

  const int t = threadIdx.x;
  const int g = t >> 4, e = t & 15;

  #pragma unroll
  for (int i = 0; i < 16; ++i) {
    sWq[t + i * 128] = Wq[t + i * 128];
    sWk[t + i * 128] = Wk[t + i * 128];
    sWv[t + i * 128] = Wv[t + i * 128];
    sWo[t + i * 128] = Wo[t + i * 128];
  }
  sbq[t] = bq[t]; sbk[t] = bk[t]; sbv[t] = bv[t]; sbo[t] = bo[t];
  __syncthreads();

  for (int s = 0; s < SPB; ++s) {
    const int n = blockIdx.x * SPB + s;
    sh[t]  = h [(size_t)n * 128 + t];
    seo[t] = eo[(size_t)n * 128 + t];
    __syncthreads();

    // top-2 gating weights (softmax over masked logits)
    float w1 = 0.f, w2 = 0.f; int i1 = 0, i2 = 0;
    if (t == 0) {
      float m1 = -1e30f;
      for (int i = 0; i < 128; ++i) if (sh[i] > m1) { m1 = sh[i]; i1 = i; }
      float m2 = -1e30f;
      for (int i = 0; i < 128; ++i) if (i != i1 && sh[i] > m2) { m2 = sh[i]; i2 = i; }
      float b  = __expf(m2 - m1);
      float iv = 1.0f / (1.0f + b);
      w1 = iv; w2 = b * iv;
    }

    // Q,K,V projections: y[g,f] = sum_j eo[g,j] * W[g,f,j] + b
    {
      float aq = sbq[t], ak = sbk[t], av = sbv[t];
      const float* wq = &sWq[t * 16];
      const float* wk = &sWk[t * 16];
      const float* wv = &sWv[t * 16];
      const float* eg = &seo[g * 16];
      #pragma unroll
      for (int j = 0; j < 16; ++j) {
        float xv = eg[j];
        aq += xv * wq[j]; ak += xv * wk[j]; av += xv * wv[j];
      }
      sq[t] = aq; sk[t] = ak; sv[t] = av;
    }
    __syncthreads();

    // scores[g,d,e2] = sum_hh Qh[g,d,hh]*Kh[g,e2,hh] / sqrt(4)
    {
      int d = e >> 2, e2 = e & 3;
      float sc = 0.f;
      #pragma unroll
      for (int hh = 0; hh < 4; ++hh)
        sc += sq[g * 16 + hh * 4 + d] * sk[g * 16 + hh * 4 + e2];
      ss[t] = sc * 0.5f;
    }
    __syncthreads();

    // softmax over last axis (e2)
    {
      int d = e >> 2;
      float mx = -1e30f;
      #pragma unroll
      for (int j = 0; j < 4; ++j) mx = fmaxf(mx, ss[g * 16 + d * 4 + j]);
      float den = 0.f;
      #pragma unroll
      for (int j = 0; j < 4; ++j) den += __expf(ss[g * 16 + d * 4 + j] - mx);
      sat[t] = __expf(ss[t] - mx) / den;
    }
    __syncthreads();

    // att_flat[g, f]  (f = hh*4 + dd):  sum_j attn[g,dd,j] * Vh[g,j,hh]
    {
      int hh = e >> 2, dd = e & 3;
      float a = 0.f;
      #pragma unroll
      for (int j = 0; j < 4; ++j)
        a += sat[g * 16 + dd * 4 + j] * sv[g * 16 + hh * 4 + j];
      saf[t] = a;
    }
    __syncthreads();

    // output projection
    {
      float a = sbo[t];
      const float* wo = &sWo[t * 16];
      const float* ag = &saf[g * 16];
      #pragma unroll
      for (int j = 0; j < 16; ++j) a += ag[j] * wo[j];
      so[t] = a;
    }
    __syncthreads();

    if (t == 0) out[n] = w1 * so[i1] + w2 * so[i2];
    __syncthreads();
  }
}

// ---------------------------------------------------------------------------
extern "C" void kernel_launch(void* const* d_in, const int* in_sizes, int n_in,
                              void* d_out, int out_size, void* d_ws, size_t ws_size,
                              hipStream_t stream) {
  const float* x  = (const float*)d_in[0];
  const float* Wr = (const float*)d_in[1];
  const float* br = (const float*)d_in[2];
  const float* We = (const float*)d_in[3];
  const float* be = (const float*)d_in[4];
  const float* Wq = (const float*)d_in[5];
  const float* bq = (const float*)d_in[6];
  const float* Wk = (const float*)d_in[7];
  const float* bk = (const float*)d_in[8];
  const float* Wv = (const float*)d_in[9];
  const float* bv = (const float*)d_in[10];
  const float* Wo = (const float*)d_in[11];
  const float* bo = (const float*)d_in[12];

  float* h  = (float*)d_ws;                               // 10.24 MB
  float* eo = h + (size_t)N_SAMP * GE_DIM;                // 10.24 MB
  unsigned short* wrt = (unsigned short*)(eo + (size_t)N_SAMP * GE_DIM); // 2.44 MB

  const int trBlocks = (GE_DIM * (KT_PAD / 2)) / 256;     // 2384, exact
  miga_wr_transpose<<<trBlocks,    256, 0, stream>>>(Wr, (unsigned int*)wrt);
  miga_router      <<<N_SAMP / MT, 256, 0, stream>>>(x, wrt, br, h);
  miga_experts     <<<N_SAMP / MT, 256, 0, stream>>>(h, We, be, eo);
  miga_attn        <<<N_SAMP / SPB,128, 0, stream>>>(h, eo, Wq, bq, Wk, bk,
                                                     Wv, bv, Wo, bo, (float*)d_out);
}
